// SignCL_12412455485837
// MI455X (gfx1250) — compile-verified, matrix operand order checked by reading, not run on hardware
//
#include <hip/hip_runtime.h>
#include <math.h>

typedef __attribute__((ext_vector_type(16))) _Float16 v16h;
typedef __attribute__((ext_vector_type(8)))  float    v8f;
typedef __attribute__((ext_vector_type(4)))  float    v4f;

namespace {
constexpr int   kB = 16, kS = 2048, kD = 512;
constexpr int   kNA = kS - 3;                       // 2045 anchors
constexpr float kMaxDist = 32.0f;
constexpr int   kWavesPerBlock = 8;                 // 256 threads, wave32
constexpr int   kTiles = (kNA + 15) / 16;           // 128 tiles of 16 anchors
constexpr int   kTileGroups = kTiles / kWavesPerBlock; // 16
constexpr int   kGrid = kTileGroups * kB;           // 256 blocks
// per-contribution weights: mean over (B, n_samples), then /(B*(S-4))
constexpr float kDen  = (float)kB * (float)(kS - 4);
constexpr float kPosW = 1.0f / (2.0f * (float)kB * kDen);
constexpr float kNegW = 1.0f / (4.0f * (float)kB * kDen);
}

__device__ __forceinline__ float softplus_f(float x) {
  // numerically stable softplus
  return fmaxf(x, 0.0f) + log1pf(__expf(-fabsf(x)));
}

__global__ __launch_bounds__(256, 2) void signcl_wmma_kernel(
    const float* __restrict__ emb, const int* __restrict__ pos_idx,
    const int* __restrict__ neg_idx, float* __restrict__ partials)
{
  const int lane = threadIdx.x & 31;
  const int wave = threadIdx.x >> 5;
  const int half = lane >> 4;       // which 16-lane half of the wave
  const int row  = lane & 15;       // A-matrix row index M

  const int b    = blockIdx.x & (kB - 1);
  const int tg   = blockIdx.x >> 4;
  const int tile = tg * kWavesPerBlock + wave;     // 0..127

  // anchor index handled by this lane's A-row (clamped for the partial tile)
  const int j  = tile * 16 + row;
  const int jc = (j < kNA) ? j : (kNA - 1);

  const float* __restrict__ arow = emb + ((size_t)b * kS + (size_t)(jc + 1)) * kD;
  const float* srow[6];
  srow[0] = emb + ((size_t)b * kS + (size_t)pos_idx[2 * jc + 0]) * kD;
  srow[1] = emb + ((size_t)b * kS + (size_t)pos_idx[2 * jc + 1]) * kD;
#pragma unroll
  for (int k = 0; k < 4; ++k)
    srow[2 + k] = emb + ((size_t)b * kS + (size_t)neg_idx[4 * jc + k]) * kD;

  v16h ones;
#pragma unroll
  for (int p = 0; p < 16; ++p) ones[p] = (_Float16)1.0f;

  v8f acc[6] = {};   // one f32 16x16 accumulator per sample

  // K loop: 16 chunks of 32 over D=512. Each lane supplies 16 elements of its
  // row per chunk ({half*8..+7} and {+16..+23} relative to chunk base), which
  // exactly tiles the chunk across the two lane halves. B = ones, so
  // D[m][*] = sum_k |a-s| regardless of intra-lane K placement.
  for (int kc = 0; kc < kD; kc += 32) {
    const int o0 = kc + half * 8;
    const v4f a0 = *(const v4f*)(arow + o0);
    const v4f a1 = *(const v4f*)(arow + o0 + 4);
    const v4f a2 = *(const v4f*)(arow + o0 + 16);
    const v4f a3 = *(const v4f*)(arow + o0 + 20);
#pragma unroll
    for (int s = 0; s < 6; ++s) {
      const float* __restrict__ sp = srow[s];
      const v4f s0 = *(const v4f*)(sp + o0);
      const v4f s1 = *(const v4f*)(sp + o0 + 4);
      const v4f s2 = *(const v4f*)(sp + o0 + 16);
      const v4f s3 = *(const v4f*)(sp + o0 + 20);
      v16h av;
#pragma unroll
      for (int p = 0; p < 4; ++p) {
        av[p +  0] = (_Float16)fabsf(a0[p] - s0[p]);
        av[p +  4] = (_Float16)fabsf(a1[p] - s1[p]);
        av[p +  8] = (_Float16)fabsf(a2[p] - s2[p]);
        av[p + 12] = (_Float16)fabsf(a3[p] - s3[p]);
      }
      acc[s] = __builtin_amdgcn_wmma_f32_16x16x32_f16(
          false, av, false, ones, (short)0, acc[s], false, false);
    }
  }

  // C/D layout: lanes 0-15 hold rows 0-7 in VGPRs 0-7 (all columns equal),
  // lanes 16-31 hold rows 8-15. Every lane of a half therefore holds all 8
  // row sums of that half -> compute softplus contributions per lane.
  float part = 0.0f;
#pragma unroll
  for (int s = 0; s < 6; ++s) {
    const float w = (s < 2) ? kPosW : kNegW;
#pragma unroll
    for (int v = 0; v < 8; ++v) {
      const int m  = v + half * 8;
      const int jj = tile * 16 + m;
      if (jj < kNA) {
        const float d = acc[s][v];
        const float x = (s < 2) ? (d - kMaxDist) : (kMaxDist - d);
        part += w * softplus_f(x);
      }
    }
  }

  // lanes within a half are identical; combine the two halves once.
  part = __shfl(part, 0, 32) + __shfl(part, 16, 32);

  __shared__ float lds[kWavesPerBlock];
  if (lane == 0) lds[wave] = part;
  __syncthreads();
  if (threadIdx.x == 0) {
    float t = 0.0f;
#pragma unroll
    for (int i = 0; i < kWavesPerBlock; ++i) t += lds[i];
    partials[blockIdx.x] = t;
  }
}

__global__ __launch_bounds__(256) void signcl_reduce_kernel(
    const float* __restrict__ partials, float* __restrict__ out, int n)
{
  __shared__ float lds[256];
  float t = 0.0f;
  for (int i = threadIdx.x; i < n; i += 256) t += partials[i];
  lds[threadIdx.x] = t;
  __syncthreads();
  for (int sft = 128; sft > 0; sft >>= 1) {
    if ((int)threadIdx.x < sft) lds[threadIdx.x] += lds[threadIdx.x + sft];
    __syncthreads();
  }
  if (threadIdx.x == 0) out[0] = lds[0];
}

extern "C" void kernel_launch(void* const* d_in, const int* in_sizes, int n_in,
                              void* d_out, int out_size, void* d_ws, size_t ws_size,
                              hipStream_t stream) {
  const float* emb     = (const float*)d_in[0];
  const int*   pos_idx = (const int*)d_in[1];
  const int*   neg_idx = (const int*)d_in[2];
  // d_in[3] = margin (unused; baked into index generation)
  float* partials = (float*)d_ws;    // kGrid floats = 1 KB scratch
  float* out      = (float*)d_out;

  signcl_wmma_kernel<<<kGrid, 256, 0, stream>>>(emb, pos_idx, neg_idx, partials);
  signcl_reduce_kernel<<<1, 256, 0, stream>>>(partials, out, kGrid);
}
